// SimDecoderLayer_2302102471072
// MI455X (gfx1250) — compile-verified, matrix operand order checked by reading, not run on hardware
//
#include <hip/hip_runtime.h>
#include <hip/hip_bf16.h>

// ---------------------------------------------------------------------------
// SimDecoderLayer on MI455X (gfx1250), WMMA f16-in/f32-acc matrix pipe.
// GEMM: each wave owns a 16x32 C tile (2 accumulators), K-loop unrolled x2
// -> 4 independent v_wmma_f32_16x16x32_f16 per iteration (no hazard nops).
// Weights are pre-packed once to f16 in WMMA B-fragment order so a B fragment
// is two aligned global_load_b128. Activations load f32 + v_cvt_pk_f16_f32.
// Template specialization removes all bounds branches in the hot shapes.
// ---------------------------------------------------------------------------

typedef __attribute__((ext_vector_type(16))) _Float16 v16h;
typedef __attribute__((ext_vector_type(8)))  _Float16 v8h;
typedef __attribute__((ext_vector_type(8)))  float    v8f;

#define GF_BT    1   // B is f32 [N,K] row-major (transposed-B GEMM)
#define GF_GM    2   // guard M (M not multiple of 32)
#define GF_GN    4   // guard N (N not multiple of 128)
#define GF_HALF  8   // store C as f16
#define GF_RELU 16   // relu epilogue

// A-fragment (16-bit A 16x32 ISA layout): lane<16 -> K {0..7,16..23},
// lane>=16 -> K {8..15,24..31}; element e -> k = (e<8?0:16) + hs*8 + (e&7).
__device__ __forceinline__ v16h load_a_frag(const float* pa /* = &A[row*lda + k0 + hs*8] */)
{
  v16h af;
  #pragma unroll
  for (int g = 0; g < 2; ++g) {
    float4 f0 = *(const float4*)(pa + g * 16);
    float4 f1 = *(const float4*)(pa + g * 16 + 4);
    af[8*g+0]=(_Float16)f0.x; af[8*g+1]=(_Float16)f0.y;
    af[8*g+2]=(_Float16)f0.z; af[8*g+3]=(_Float16)f0.w;
    af[8*g+4]=(_Float16)f1.x; af[8*g+5]=(_Float16)f1.y;
    af[8*g+6]=(_Float16)f1.z; af[8*g+7]=(_Float16)f1.w;
  }
  return af;
}

// Packed-B fragment: Bp layout [K/32][N][32] f16; two aligned b128 loads.
__device__ __forceinline__ v16h load_bp_frag(const _Float16* p /* 32-elem k-block for col */, int hs)
{
  v8h c0 = *(const v8h*)(p + hs * 8);
  v8h c1 = *(const v8h*)(p + 16 + hs * 8);
  return __builtin_shufflevector(c0, c1, 0,1,2,3,4,5,6,7,8,9,10,11,12,13,14,15);
}

template <int F>
__global__ __launch_bounds__(256) void wmma_gemm(
    const float* __restrict__ A, int lda, long long sAb, long long sAh,
    const void* __restrict__ Bv, int ldb, long long sBb, long long sBh,
    void* __restrict__ Cv, int ldc, long long sCb, long long sCh,
    const float* __restrict__ bias, int M, int N, int K)
{
  constexpr bool BT   = (F & GF_BT)   != 0;
  constexpr bool GM   = (F & GF_GM)   != 0;
  constexpr bool GN   = (F & GF_GN)   != 0;
  constexpr bool HALF = (F & GF_HALF) != 0;
  constexpr bool RELU = (F & GF_RELU) != 0;

  const int wave = threadIdx.x >> 5;
  const int lane = threadIdx.x & 31;
  const int hs   = lane >> 4;
  const int lr   = lane & 15;
  const int mt   = blockIdx.y * 2 + (wave >> 2);
  const int ntb  = (blockIdx.x * 4 + (wave & 3)) * 2;   // first of 2 N-tiles
  const long long bb = (long long)(blockIdx.z >> 1);
  const long long hh = (long long)(blockIdx.z & 1);
  A += bb * sAb + hh * sAh;
  const long long boff = bb * sBb + hh * sBh;
  const long long coff = bb * sCb + hh * sCh;

  const int row  = mt * 16 + lr;
  const int col0 = ntb * 16 + lr;
  const int col1 = col0 + 16;
  const bool rowOK  = !GM || (row  < M);
  const bool col0OK = !GN || (col0 < N);
  const bool col1OK = !GN || (col1 < N);

  v8f acc0 = {};
  v8f acc1 = {};

  #pragma unroll 2
  for (int k0 = 0; k0 < K; k0 += 32) {
    v16h af = {};
    if (rowOK) {
      const float* pa = A + (size_t)row * lda + k0 + hs * 8;
      __builtin_prefetch(pa + 64, 0, 1);        // global_prefetch_b8 ahead in K
      af = load_a_frag(pa);
    }
    v16h bf0 = {}, bf1 = {};
    if (BT) {
      const float* Bf = (const float*)Bv + boff;
      if (col0OK) bf0 = load_a_frag(Bf + (size_t)col0 * ldb + k0 + hs * 8);
      if (col1OK) bf1 = load_a_frag(Bf + (size_t)col1 * ldb + k0 + hs * 8);
    } else {
      const _Float16* Bp = (const _Float16*)Bv;  // [K/32][N][32] packed f16
      size_t kb = (size_t)(k0 >> 5) * N;
      if (col0OK) bf0 = load_bp_frag(Bp + (kb + col0) * 32, hs);
      if (col1OK) bf1 = load_bp_frag(Bp + (kb + col1) * 32, hs);
    }
    // EXEC is full at the WMMAs (structured reconvergence, no early returns).
    acc0 = __builtin_amdgcn_wmma_f32_16x16x32_f16(false, af, false, bf0, (short)0, acc0, false, false);
    acc1 = __builtin_amdgcn_wmma_f32_16x16x32_f16(false, af, false, bf1, (short)0, acc1, false, false);
  }

  // C/D layout: VGPR r -> row = mt*16 + hs*8 + r, col = lane&15 within tile.
  const int orow0 = mt * 16 + hs * 8;
  #pragma unroll
  for (int t = 0; t < 2; ++t) {
    const int col = (t == 0) ? col0 : col1;
    const bool cOK = (t == 0) ? col0OK : col1OK;
    if (cOK) {
      float bv = bias ? bias[col] : 0.0f;
      #pragma unroll
      for (int r = 0; r < 8; ++r) {
        int orow = orow0 + r;
        if (!GM || (orow < M)) {
          float v = ((t == 0) ? acc0[r] : acc1[r]) + bv;
          if (RELU) v = v > 0.0f ? v : 0.0f;
          size_t ci = (size_t)orow * ldc + col + (size_t)coff;
          if (HALF) ((_Float16*)Cv)[ci] = (_Float16)v;
          else      ((float*)Cv)[ci]    = v;
        }
      }
    }
  }
}

// pack f32 weight B[K,N] -> f16 Bp[K/32][N][32] (WMMA fragment order)
__global__ void k_pack_b(const float* __restrict__ B, _Float16* __restrict__ Bp,
                         int K, int N)
{
  int t = blockIdx.x * blockDim.x + threadIdx.x;
  if (t >= K * N) return;
  int k = t / N, n = t % N;
  Bp[((size_t)(k >> 5) * N + n) * 32 + (k & 31)] = (_Float16)B[t];
}

// ---------------------------------------------------------------------------
__global__ void k_add(const float* __restrict__ a, const float* __restrict__ b,
                      float* __restrict__ c, int n)
{
  int t = blockIdx.x * blockDim.x + threadIdx.x;
  if (t < n) c[t] = a[t] + b[t];
}

// softmax over groups of 16 (N_LEVELS*N_POINTS) within rows of 128 (8 heads)
__global__ void k_softmax16(float* __restrict__ attn, int ngroups)
{
  int t = blockIdx.x * blockDim.x + threadIdx.x;
  if (t >= ngroups) return;
  float* p = attn + (size_t)t * 16;
  float m = p[0];
  #pragma unroll
  for (int j = 1; j < 16; ++j) m = fmaxf(m, p[j]);
  float e[16]; float s = 0.0f;
  #pragma unroll
  for (int j = 0; j < 16; ++j) { e[j] = __expf(p[j] - m); s += e[j]; }
  float inv = 1.0f / s;
  #pragma unroll
  for (int j = 0; j < 16; ++j) p[j] = e[j] * inv;
}

// Deformable sampling: one wave per (b, q, head); lane == channel d (0..31).
// value stored f16 as [b, lv, head, d] so a wave's gather is 64 contiguous B.
__global__ void k_sample(const _Float16* __restrict__ val,
                         const float* __restrict__ refpts,
                         const float* __restrict__ offb,
                         const float* __restrict__ attnb,
                         float* __restrict__ samp)
{
  const int gw   = (blockIdx.x * blockDim.x + threadIdx.x) >> 5;
  const int lane = threadIdx.x & 31;
  if (gw >= 8 * 300 * 8) return;
  const int h  = gw & 7;
  const int bq = gw >> 3;
  const int q  = bq % 300;
  const int b  = bq / 300;

  const int lH[4]  = {128, 64, 32, 16};
  const int lW[4]  = {128, 64, 32, 16};
  const int lS0[4] = {0, 16384, 20480, 21504};

  float acc = 0.0f;
  #pragma unroll
  for (int l = 0; l < 4; ++l) {
    const int H_ = lH[l], W_ = lW[l], s0 = lS0[l];
    const float rx = refpts[((size_t)(b * 300 + q) * 4 + l) * 2 + 0];
    const float ry = refpts[((size_t)(b * 300 + q) * 4 + l) * 2 + 1];
    #pragma unroll
    for (int p = 0; p < 4; ++p) {
      const int o = (((b * 300 + q) * 8 + h) * 4 + l) * 4 + p;
      const float ox = offb[(size_t)o * 2 + 0];
      const float oy = offb[(size_t)o * 2 + 1];
      const float aw = attnb[(size_t)(b * 300 + q) * 128 + h * 16 + l * 4 + p];
      // loc = ref + off/(W,H); x = loc.x*W - 0.5 => ref.x*W + off.x - 0.5
      const float x = rx * (float)W_ + ox - 0.5f;
      const float y = ry * (float)H_ + oy - 0.5f;
      const float x0f = floorf(x), y0f = floorf(y);
      const float lx = x - x0f, ly = y - y0f;
      const int x0 = (int)x0f, y0 = (int)y0f;
      float w[4]  = {(1.0f-lx)*(1.0f-ly), lx*(1.0f-ly), (1.0f-lx)*ly, lx*ly};
      int   yy[4] = {y0, y0, y0+1, y0+1};
      int   xx[4] = {x0, x0+1, x0, x0+1};
      #pragma unroll
      for (int cnr = 0; cnr < 4; ++cnr) {
        bool valid = (yy[cnr] >= 0) & (yy[cnr] < H_) & (xx[cnr] >= 0) & (xx[cnr] < W_);
        int yc = min(max(yy[cnr], 0), H_ - 1);
        int xc = min(max(xx[cnr], 0), W_ - 1);
        int lv = s0 + yc * W_ + xc;
        float v = (float)val[((size_t)(b * 21760 + lv)) * 256 + h * 32 + lane];
        acc += aw * (valid ? w[cnr] : 0.0f) * v;
      }
    }
  }
  samp[(size_t)(b * 300 + q) * 256 + h * 32 + lane] = acc;
}

// LayerNorm(x + r) * g + b; one wave per 256-element row (wave32: 8/lane)
__global__ void k_ln(const float* __restrict__ x, const float* __restrict__ r,
                     const float* __restrict__ g, const float* __restrict__ bta,
                     float* __restrict__ out, int rows)
{
  const int wid  = (blockIdx.x * blockDim.x + threadIdx.x) >> 5;
  const int lane = threadIdx.x & 31;
  if (wid >= rows) return;               // uniform per wave
  const size_t base = (size_t)wid * 256 + lane * 8;
  float v[8]; float s = 0.0f;
  #pragma unroll
  for (int i = 0; i < 8; ++i) { v[i] = x[base + i] + r[base + i]; s += v[i]; }
  #pragma unroll
  for (int o = 16; o > 0; o >>= 1) s += __shfl_xor(s, o, 32);
  const float m = s * (1.0f / 256.0f);
  float vs = 0.0f;
  #pragma unroll
  for (int i = 0; i < 8; ++i) { float d = v[i] - m; vs += d * d; }
  #pragma unroll
  for (int o = 16; o > 0; o >>= 1) vs += __shfl_xor(vs, o, 32);
  const float rstd = rsqrtf(vs * (1.0f / 256.0f) + 1e-5f);
  #pragma unroll
  for (int i = 0; i < 8; ++i) {
    int c = lane * 8 + i;
    out[base + i] = (v[i] - m) * rstd * g[c] + bta[c];
  }
}

// per-(row, head) inverse L2 norm over 128 channels: o[row*2+h]
__global__ void k_rownorm(const float* __restrict__ x, float* __restrict__ o, int total)
{
  int t = blockIdx.x * blockDim.x + threadIdx.x;
  if (t >= total) return;
  int row = t >> 1, h = t & 1;
  const float* p = x + (size_t)row * 256 + h * 128;
  float s = 0.0f;
  for (int i = 0; i < 128; ++i) { float v = p[i]; s += v * v; }
  o[t] = rsqrtf(s + 1e-12f);
}

// head-mean of scaled dot product and of cosine similarity
__global__ void k_epi(const float* __restrict__ S, const float* __restrict__ rq,
                      const float* __restrict__ rk, float* __restrict__ ocos,
                      float* __restrict__ omm, int NK, int total)
{
  int t = blockIdx.x * blockDim.x + threadIdx.x;
  if (t >= total) return;
  int kk  = t % NK;
  int rem = t / NK;
  int q   = rem % 300;
  int b   = rem / 300;
  size_t i0 = ((size_t)(b * 2 + 0) * 300 + q) * NK + kk;
  size_t i1 = ((size_t)(b * 2 + 1) * 300 + q) * NK + kk;
  float s0 = S[i0], s1 = S[i1];
  float rq0 = rq[(b * 300 + q) * 2 + 0], rq1 = rq[(b * 300 + q) * 2 + 1];
  float rk0 = rk[(b * NK + kk) * 2 + 0], rk1 = rk[(b * NK + kk) * 2 + 1];
  omm[t]  = 0.5f * (s0 + s1) * 0.08838834764831845f;   // * 128^-0.5, head mean
  ocos[t] = 0.5f * (s0 * rq0 * rk0 + s1 * rq1 * rk1);
}

// ---------------------------------------------------------------------------
template <int F>
static void launch_gemm(const float* A, int lda, long long sAb, long long sAh,
                        const void* B, int ldb, long long sBb, long long sBh,
                        void* C, int ldc, long long sCb, long long sCh,
                        const float* bias, int M, int N, int K, int batches,
                        hipStream_t stream)
{
  dim3 grid((N + 127) / 128, (M + 31) / 32, batches);
  wmma_gemm<F><<<grid, dim3(256), 0, stream>>>(A, lda, sAb, sAh, B, ldb, sBb, sBh,
                                               C, ldc, sCb, sCh, bias, M, N, K);
}

extern "C" void kernel_launch(void* const* d_in, const int* in_sizes, int n_in,
                              void* d_out, int out_size, void* d_ws, size_t ws_size,
                              hipStream_t stream)
{
  (void)in_sizes; (void)n_in; (void)out_size; (void)ws_size;
  const float* proposals = (const float*)d_in[0];
  const float* track     = (const float*)d_in[1];
  const float* ppos      = (const float*)d_in[2];
  const float* refpts    = (const float*)d_in[4];
  const float* src       = (const float*)d_in[5];
  const float* Wv  = (const float*)d_in[9];  const float* bv  = (const float*)d_in[10];
  const float* Wsw = (const float*)d_in[11]; const float* bsw = (const float*)d_in[12];
  const float* Wa  = (const float*)d_in[13]; const float* ba  = (const float*)d_in[14];
  const float* Wo  = (const float*)d_in[15]; const float* bo  = (const float*)d_in[16];
  const float* g1  = (const float*)d_in[17]; const float* b1  = (const float*)d_in[18];
  const float* W1  = (const float*)d_in[19]; const float* bf1 = (const float*)d_in[20];
  const float* W2  = (const float*)d_in[21]; const float* bf2 = (const float*)d_in[22];
  const float* g3  = (const float*)d_in[23]; const float* b3  = (const float*)d_in[24];
  const float* Wq  = (const float*)d_in[25]; const float* bq  = (const float*)d_in[26];
  const float* Wk  = (const float*)d_in[27]; const float* bk  = (const float*)d_in[28];

  // ---- workspace carve-up (256B aligned) ----
  char* wsb = (char*)d_ws;
  size_t off = 0;
  auto take = [&](size_t bytes) -> char* {
    char* p = wsb + off;
    off = (off + bytes + 255) & ~(size_t)255;
    return p;
  };
  _Float16* val_h = (_Float16*)take(174080ull * 256 * 2);  // 89.1 MB f16 value
  float* qbuf  = (float*)take(614400ull * 4);
  float* offb  = (float*)take(614400ull * 4);
  float* attnb = (float*)take(307200ull * 4);
  float* sampb = (float*)take(614400ull * 4);
  float* tmpb  = (float*)take(614400ull * 4);   // msda out / FFN down out
  float* prop1 = (float*)take(614400ull * 4);
  float* ffnb  = (float*)take(2457600ull * 4);
  float* qh    = (float*)take(614400ull * 4);
  float* kht   = (float*)take(102400ull * 4);
  float* khp   = (float*)take(614400ull * 4);
  float* Spt   = (float*)take(240000ull * 4);
  float* Spp   = (float*)take(1440000ull * 4);
  float* rqv   = (float*)take(4800ull * 4);
  float* rkt   = (float*)take(800ull * 4);
  float* rkp   = (float*)take(4800ull * 4);
  // packed f16 weights (WMMA B-fragment order)
  _Float16* Wv_p = (_Float16*)take(65536ull * 2);
  _Float16* Ws_p = (_Float16*)take(65536ull * 2);
  _Float16* Wa_p = (_Float16*)take(32768ull * 2);
  _Float16* Wo_p = (_Float16*)take(65536ull * 2);
  _Float16* W1_p = (_Float16*)take(262144ull * 2);
  _Float16* W2_p = (_Float16*)take(262144ull * 2);
  _Float16* Wq_p = (_Float16*)take(65536ull * 2);
  _Float16* Wk_p = (_Float16*)take(65536ull * 2);

  float* out_prop  = (float*)d_out;               // [8,300,256]
  float* out_ptcos = out_prop + 614400;           // [8,300,50]
  float* out_ptmm  = out_ptcos + 120000;
  float* out_ppcos = out_ptmm + 120000;           // [8,300,300]
  float* out_ppmm  = out_ppcos + 720000;

  // 0) pack all weights to f16 fragment order
  auto pack = [&](const float* B, _Float16* Bp, int K, int N) {
    int n = K * N;
    k_pack_b<<<dim3((n + 255) / 256), dim3(256), 0, stream>>>(B, Bp, K, N);
  };
  pack(Wv, Wv_p, 256, 256);   pack(Wsw, Ws_p, 256, 256);
  pack(Wa, Wa_p, 256, 128);   pack(Wo, Wo_p, 256, 256);
  pack(W1, W1_p, 256, 1024);  pack(W2, W2_p, 1024, 256);
  pack(Wq, Wq_p, 256, 256);   pack(Wk, Wk_p, 256, 256);

  // 1) q = proposals + p_query_pos
  k_add<<<dim3(2400), dim3(256), 0, stream>>>(proposals, ppos, qbuf, 614400);

  // 2) value = src @ Wv + bv  (M=174080, N=256, K=256) -> f16, no guards
  launch_gemm<GF_HALF>(src, 256, 0, 0, Wv_p, 0, 0, 0, val_h, 256, 0, 0, bv,
                       174080, 256, 256, 1, stream);

  // 3) sampling offsets = q @ Ws + bs
  launch_gemm<0>(qbuf, 256, 0, 0, Ws_p, 0, 0, 0, offb, 256, 0, 0, bsw,
                 2400, 256, 256, 1, stream);

  // 4) attention logits = q @ Wa + ba, then softmax over 16
  launch_gemm<0>(qbuf, 256, 0, 0, Wa_p, 0, 0, 0, attnb, 128, 0, 0, ba,
                 2400, 128, 256, 1, stream);
  k_softmax16<<<dim3(75), dim3(256), 0, stream>>>(attnb, 19200);

  // 5) bilinear gather + weighted sum
  k_sample<<<dim3(2400), dim3(256), 0, stream>>>(val_h, refpts, offb, attnb, sampb);

  // 6) output proj + residual LN1
  launch_gemm<0>(sampb, 256, 0, 0, Wo_p, 0, 0, 0, tmpb, 256, 0, 0, bo,
                 2400, 256, 256, 1, stream);
  k_ln<<<dim3(300), dim3(256), 0, stream>>>(proposals, tmpb, g1, b1, prop1, 2400);

  // 7) FFN: relu(prop1 @ W1 + bf1) @ W2 + bf2, residual LN3 -> d_out
  launch_gemm<GF_RELU>(prop1, 256, 0, 0, W1_p, 0, 0, 0, ffnb, 1024, 0, 0, bf1,
                       2400, 1024, 256, 1, stream);
  launch_gemm<0>(ffnb, 1024, 0, 0, W2_p, 0, 0, 0, tmpb, 256, 0, 0, bf2,
                 2400, 256, 1024, 1, stream);
  k_ln<<<dim3(300), dim3(256), 0, stream>>>(prop1, tmpb, g3, b3, out_prop, 2400);

  // 8) weight-attention projections
  launch_gemm<0>(out_prop, 256, 0, 0, Wq_p, 0, 0, 0, qh, 256, 0, 0, bq,
                 2400, 256, 256, 1, stream);
  launch_gemm<GF_GM>(track, 256, 0, 0, Wk_p, 0, 0, 0, kht, 256, 0, 0, bk,
                     400, 256, 256, 1, stream);
  launch_gemm<0>(out_prop, 256, 0, 0, Wk_p, 0, 0, 0, khp, 256, 0, 0, bk,
                 2400, 256, 256, 1, stream);

  // 9) per-(row,head) inverse norms
  k_rownorm<<<dim3(19), dim3(256), 0, stream>>>(qh, rqv, 4800);
  k_rownorm<<<dim3(4),  dim3(256), 0, stream>>>(kht, rkt, 800);
  k_rownorm<<<dim3(19), dim3(256), 0, stream>>>(khp, rkp, 4800);

  // 10) S[b,h] = qh_h @ kh_h^T (batched over grid.z = 16 = 8 batches x 2 heads)
  launch_gemm<GF_BT | GF_GM | GF_GN>(qh, 256, 76800, 128, kht, 256, 12800, 128,
                                     Spt, 50, 30000, 15000, nullptr,
                                     300, 50, 128, 16, stream);
  launch_gemm<GF_BT | GF_GM | GF_GN>(qh, 256, 76800, 128, khp, 256, 76800, 128,
                                     Spp, 300, 180000, 90000, nullptr,
                                     300, 300, 128, 16, stream);

  // 11) head-mean cos / scaled-dot outputs
  k_epi<<<dim3(469),  dim3(256), 0, stream>>>(Spt, rqv, rkt, out_ptcos, out_ptmm, 50, 120000);
  k_epi<<<dim3(2813), dim3(256), 0, stream>>>(Spp, rqv, rkp, out_ppcos, out_ppmm, 300, 720000);
}
